// Neural3DHMM_5196910428669
// MI455X (gfx1250) — matrix operand order repaired and checked by reading, not compiled, not run on hardware
//
#include <hip/hip_runtime.h>
#include <hip/hip_bf16.h>

// ---------------------------------------------------------------------------
// Neural 3D HMM forward on MI455X (gfx1250, wave32, WMMA).
// S=1024 states, EMB=256, V=10000 tokens, B=64, T=32, L=16. Output (T,B,S) f32.
// All GEMMs on v_wmma_f32_16x16x32_bf16; B operands and scan P pre-packed into the
// ISA 7.12.2 lane layouts, other A operands read as bf16 rows; each fragment is two 16-byte vector loads.
// ---------------------------------------------------------------------------

#define NS   1024
#define EMBD 256
#define NV   10000
#define NB   64
#define NT   32
#define NL   16
#define NCHUNK 25          // vocab chunks for emission lse (625 tiles = 25*25)
#define TILES_PER_CHUNK 25

typedef __attribute__((ext_vector_type(16))) __bf16 v16bf;
typedef __attribute__((ext_vector_type(8)))  __bf16 v8bf;
typedef __attribute__((ext_vector_type(8)))  float  v8f;

#define FRAG_ELEMS 512     // one 16x32(or 32x16) bf16 fragment tile = 32 lanes * 16

// ---- WMMA wrapper ---------------------------------------------------------
__device__ __forceinline__ v8f wmma_bf16(v16bf a, v16bf b, v8f c) {
  return __builtin_amdgcn_wmma_f32_16x16x32_bf16(
      false, a, false, b, (short)0, c, false, false);
}

// ---- Fragment loaders (ISA 7.12.2, 16-bit layouts, wave32) ----------------
// A 16x32: lane holds row m=lane&15; element j -> k = 16*(j>>3)+8*(lane>>4)+(j&7)
// => per lane two contiguous 8-element runs at k = 8*(lane>>4) and +16.
__device__ __forceinline__ v16bf load_a_bf16row(const __bf16* src, int ld) {
  int lane = threadIdx.x & 31;
  const __bf16* p = src + (size_t)(lane & 15) * ld + ((lane >> 4) << 3);
  v8bf lo = *(const v8bf*)p;
  v8bf hi = *(const v8bf*)(p + 16);
  return __builtin_shufflevector(lo, hi, 0,1,2,3,4,5,6,7,8,9,10,11,12,13,14,15);
}
// Fragment from a packed tile (32 lanes x 16 contiguous bf16 per lane).
__device__ __forceinline__ v16bf load_frag_packed(const __bf16* __restrict__ tile) {
  const __bf16* p = tile + (size_t)(threadIdx.x & 31) * 16;
  v8bf lo = *(const v8bf*)p;
  v8bf hi = *(const v8bf*)(p + 8);
  return __builtin_shufflevector(lo, hi, 0,1,2,3,4,5,6,7,8,9,10,11,12,13,14,15);
}

// ---------------------------------------------------------------------------
// Elementwise f32 -> bf16 copy.
// ---------------------------------------------------------------------------
__global__ __launch_bounds__(256)
void k_cvt_bf16(const float* __restrict__ in, __bf16* __restrict__ out) {
  size_t i = (size_t)blockIdx.x * 256 + threadIdx.x;
  out[i] = (__bf16)in[i];
}

// ---------------------------------------------------------------------------
// Pack an [R x 256] f32 row-major matrix into B^T fragment tiles:
// B[k][n] = in[(nt*16+n)*256 + kt*32 + k];  tile (nt,kt) -> 512 bf16.
// grid.x = R, 256 threads.
// ---------------------------------------------------------------------------
__global__ __launch_bounds__(256)
void k_pack_bT(const float* __restrict__ in, __bf16* __restrict__ out) {
  int v = blockIdx.x, c = threadIdx.x;
  int vt = v >> 4, n = v & 15, kt = c >> 5, kin = c & 31;
  int lane = n + (((kin >> 3) & 1) << 4), j = ((kin >> 4) << 3) | (kin & 7);
  out[(((size_t)vt * 8 + kt) * 32 + lane) * 16 + j] = (__bf16)in[(size_t)v * EMBD + c];
}

// ---------------------------------------------------------------------------
// Residual layer: x1=relu(XW1^T+b1); x2=relu(x1W2^T+b2); LN(x1+x2)
// One wave / 16 rows. Weights pre-packed, X pre-converted to bf16 rows.
// ---------------------------------------------------------------------------
__global__ __launch_bounds__(32)
void k_residual(const __bf16* __restrict__ Xb,     // 1024x256 bf16 row-major
                const __bf16* __restrict__ pw1,    // w1^T packed tiles
                const float*  __restrict__ b1,
                const __bf16* __restrict__ pw2,    // w2^T packed tiles
                const float*  __restrict__ b2,
                const float*  __restrict__ g, const float* __restrict__ beta,
                float* __restrict__ out, __bf16* __restrict__ out_bf) {
  __shared__ float  lds_hf[16 * EMBD];
  __shared__ float  lds_x1f[16 * EMBD];
  __shared__ __bf16 lds_x1b[16 * EMBD];
  const int row0 = blockIdx.x * 16;
  const int lane = threadIdx.x;
  const int n_in = lane & 15, mh = (lane >> 4) << 3;

  v16bf afr[8];
#pragma unroll
  for (int kt = 0; kt < 8; ++kt)
    afr[kt] = load_a_bf16row(Xb + (size_t)row0 * EMBD + kt * 32, EMBD);

  // x1 = relu(X @ W1^T + b1)
  for (int nt = 0; nt < 16; ++nt) {
    v8f acc0 = {}, acc1 = {};
#pragma unroll
    for (int kt = 0; kt < 8; kt += 2) {
      acc0 = wmma_bf16(afr[kt],     load_frag_packed(pw1 + ((size_t)nt * 8 + kt) * FRAG_ELEMS),     acc0);
      acc1 = wmma_bf16(afr[kt + 1], load_frag_packed(pw1 + ((size_t)nt * 8 + kt + 1) * FRAG_ELEMS), acc1);
    }
    int n = nt * 16 + n_in;
    float bias = b1[n];
#pragma unroll
    for (int r = 0; r < 8; ++r) {
      float v = acc0[r] + acc1[r] + bias;
      v = v > 0.f ? v : 0.f;
      lds_x1f[(mh + r) * EMBD + n] = v;
      lds_x1b[(mh + r) * EMBD + n] = (__bf16)v;
    }
  }
  __syncthreads();

  // A fragments for GEMM-2 from the LDS bf16 copy of x1 (hoisted).
#pragma unroll
  for (int kt = 0; kt < 8; ++kt)
    afr[kt] = load_a_bf16row(lds_x1b + kt * 32, EMBD);

  // h = relu(x1 @ W2^T + b2) + x1
  for (int nt = 0; nt < 16; ++nt) {
    v8f acc0 = {}, acc1 = {};
#pragma unroll
    for (int kt = 0; kt < 8; kt += 2) {
      acc0 = wmma_bf16(afr[kt],     load_frag_packed(pw2 + ((size_t)nt * 8 + kt) * FRAG_ELEMS),     acc0);
      acc1 = wmma_bf16(afr[kt + 1], load_frag_packed(pw2 + ((size_t)nt * 8 + kt + 1) * FRAG_ELEMS), acc1);
    }
    int n = nt * 16 + n_in;
    float bias = b2[n];
#pragma unroll
    for (int r = 0; r < 8; ++r) {
      float v = acc0[r] + acc1[r] + bias;
      v = v > 0.f ? v : 0.f;
      lds_hf[(mh + r) * EMBD + n] = v + lds_x1f[(mh + r) * EMBD + n];
    }
  }
  __syncthreads();

  // LayerNorm per row (lanes 0..15 each own a row)
  if (lane < 16) {
    const float* h = lds_hf + lane * EMBD;
    float mu = 0.f;
    for (int c = 0; c < EMBD; ++c) mu += h[c];
    mu *= (1.f / EMBD);
    float var = 0.f;
    for (int c = 0; c < EMBD; ++c) { float d = h[c] - mu; var += d * d; }
    var *= (1.f / EMBD);
    float rs = rsqrtf(var + 1e-5f);
    for (int c = 0; c < EMBD; ++c) {
      float v = (h[c] - mu) * rs * g[c] + beta[c];
      out[(size_t)(row0 + lane) * EMBD + c]    = v;
      out_bf[(size_t)(row0 + lane) * EMBD + c] = (__bf16)v;
    }
  }
}

// ---------------------------------------------------------------------------
// priors = log_softmax(h_start @ start_w^T + start_b). 1 block, 1024 threads.
// ---------------------------------------------------------------------------
__global__ __launch_bounds__(1024)
void k_priors(const float* __restrict__ hs, const float* __restrict__ w,
              const float* __restrict__ b0, float* __restrict__ priors) {
  __shared__ float red[NS];
  int s = threadIdx.x;
  float logit = b0[0];
  for (int c = 0; c < EMBD; ++c) logit += hs[s * EMBD + c] * w[c];
  red[s] = logit;
  __syncthreads();
  for (int off = 512; off; off >>= 1) {
    if (s < off) red[s] = fmaxf(red[s], red[s + off]);
    __syncthreads();
  }
  float mx = red[0];
  __syncthreads();
  red[s] = expf(logit - mx);
  __syncthreads();
  for (int off = 512; off; off >>= 1) {
    if (s < off) red[s] += red[s + off];
    __syncthreads();
  }
  priors[s] = logit - (mx + logf(red[0]));
}

// ---------------------------------------------------------------------------
// logits(1024x1024) = h_in @ h_out^T.  A: bf16 rows, B: packed tiles.
// grid (64,64), 1 wave.
// ---------------------------------------------------------------------------
__global__ __launch_bounds__(32)
void k_trans_logits(const __bf16* __restrict__ Ab, const __bf16* __restrict__ Bp,
                    float* __restrict__ C) {
  const int m0 = blockIdx.x * 16, nt = blockIdx.y;
  const int lane = threadIdx.x;
  v8f acc0 = {}, acc1 = {};
#pragma unroll
  for (int kt = 0; kt < 8; kt += 2) {
    acc0 = wmma_bf16(load_a_bf16row(Ab + (size_t)m0 * EMBD + kt * 32, EMBD),
                     load_frag_packed(Bp + ((size_t)nt * 8 + kt) * FRAG_ELEMS), acc0);
    acc1 = wmma_bf16(load_a_bf16row(Ab + (size_t)m0 * EMBD + (kt + 1) * 32, EMBD),
                     load_frag_packed(Bp + ((size_t)nt * 8 + kt + 1) * FRAG_ELEMS), acc1);
  }
  int n = lane & 15, mh = (lane >> 4) << 3;
#pragma unroll
  for (int r = 0; r < 8; ++r)
    C[(size_t)(m0 + mh + r) * NS + nt * 16 + n] = acc0[r] + acc1[r];
}

// ---------------------------------------------------------------------------
// Row logsumexp over 1024 cols. 1 block/row, 256 threads.
// ---------------------------------------------------------------------------
__global__ __launch_bounds__(256)
void k_row_lse(const float* __restrict__ M, float* __restrict__ lse) {
  __shared__ float red[256];
  int r = blockIdx.x, t = threadIdx.x;
  const float* row = M + (size_t)r * NS;
  float mx = -3.0e38f;
  for (int c = t; c < NS; c += 256) mx = fmaxf(mx, row[c]);
  red[t] = mx;
  __syncthreads();
  for (int off = 128; off; off >>= 1) {
    if (t < off) red[t] = fmaxf(red[t], red[t + off]);
    __syncthreads();
  }
  float M0 = red[0];
  __syncthreads();
  float se = 0.f;
  for (int c = t; c < NS; c += 256) se += expf(row[c] - M0);
  red[t] = se;
  __syncthreads();
  for (int off = 128; off; off >>= 1) {
    if (t < off) red[t] += red[t + off];
    __syncthreads();
  }
  if (t == 0) lse[r] = M0 + logf(red[0]);
}

// ---------------------------------------------------------------------------
// c_dst[d] = max_src(logits[src,d] - row_lse[src]).  4 blocks x 256.
// ---------------------------------------------------------------------------
__global__ __launch_bounds__(256)
void k_col_max(const float* __restrict__ logits, const float* __restrict__ rl,
               float* __restrict__ cmax) {
  int d = blockIdx.x * 256 + threadIdx.x;
  float m = -3.0e38f;
  for (int s = 0; s < NS; ++s) m = fmaxf(m, logits[(size_t)s * NS + d] - rl[s]);
  cmax[d] = m;
}

// ---------------------------------------------------------------------------
// Qp packed: tile(kt,nt) lane/j layout, Q = exp(logits - row_lse - cmax) bf16.
// ---------------------------------------------------------------------------
__global__ __launch_bounds__(256)
void k_qexp(const float* __restrict__ logits, const float* __restrict__ rl,
            const float* __restrict__ cmax, __bf16* __restrict__ Qp) {
  size_t i = (size_t)blockIdx.x * 256 + threadIdx.x;
  int src = (int)(i >> 10), dst = (int)(i & 1023);
  float q = expf(logits[i] - rl[src] - cmax[dst]);
  int kt = src >> 5, kin = src & 31, nt2 = dst >> 4, n = dst & 15;
  int lane = n + (((kin >> 3) & 1) << 4), j = ((kin >> 4) << 3) | (kin & 7);
  Qp[(((size_t)kt * 64 + nt2) * 32 + lane) * 16 + j] = (__bf16)q;
}

// ---------------------------------------------------------------------------
// G_bf[t*64+b, :] = (bf16) sum_l tok_emb[stories[b,t,l], :]. 2048 blocks x 256.
// ---------------------------------------------------------------------------
__global__ __launch_bounds__(256)
void k_gather(const int* __restrict__ stories, const float* __restrict__ tok_emb,
              __bf16* __restrict__ G_bf) {
  int tb = blockIdx.x;
  int t = tb >> 6, b = tb & 63;
  int c = threadIdx.x;
  const int* toks = stories + ((size_t)b * NT + t) * NL;
  float acc = 0.f;
#pragma unroll
  for (int l = 0; l < NL; ++l) acc += tok_emb[(size_t)toks[l] * EMBD + c];
  G_bf[(size_t)tb * EMBD + c] = (__bf16)acc;
}

// ---------------------------------------------------------------------------
// Emission lse partials: chunk of 25 vocab tiles, online max/sumexp per state.
// grid (64, 25), 1 wave.
// ---------------------------------------------------------------------------
__global__ __launch_bounds__(32)
void k_emit_lse_partial(const __bf16* __restrict__ hb_emit,
                        const __bf16* __restrict__ pk_tok,
                        float* __restrict__ pm, float* __restrict__ ps) {
  __shared__ float tile[16 * 16];
  const int s0 = blockIdx.x * 16;
  const int chunk = blockIdx.y;
  const int lane = threadIdx.x;
  const int n = lane & 15, mh = (lane >> 4) << 3;

  v16bf afr[8];
#pragma unroll
  for (int kt = 0; kt < 8; ++kt)
    afr[kt] = load_a_bf16row(hb_emit + (size_t)s0 * EMBD + kt * 32, EMBD);

  float m_run = -3.0e38f, s_run = 0.f;
  const int vt0 = chunk * TILES_PER_CHUNK;
  for (int vt = vt0; vt < vt0 + TILES_PER_CHUNK; ++vt) {
    v8f acc0 = {}, acc1 = {};
#pragma unroll
    for (int kt = 0; kt < 8; kt += 2) {
      acc0 = wmma_bf16(afr[kt],     load_frag_packed(pk_tok + ((size_t)vt * 8 + kt) * FRAG_ELEMS),     acc0);
      acc1 = wmma_bf16(afr[kt + 1], load_frag_packed(pk_tok + ((size_t)vt * 8 + kt + 1) * FRAG_ELEMS), acc1);
    }
#pragma unroll
    for (int r = 0; r < 8; ++r) tile[(mh + r) * 16 + n] = acc0[r] + acc1[r];
    __syncthreads();
    if (lane < 16) {
      const float* row = tile + lane * 16;
      float tm = row[0];
      for (int j = 1; j < 16; ++j) tm = fmaxf(tm, row[j]);
      if (tm > m_run) { s_run *= expf(m_run - tm); m_run = tm; }
      for (int j = 0; j < 16; ++j) s_run += expf(row[j] - m_run);
    }
    __syncthreads();
  }
  if (lane < 16) {
    pm[(size_t)chunk * NS + s0 + lane] = m_run;
    ps[(size_t)chunk * NS + s0 + lane] = s_run;
  }
}

__global__ __launch_bounds__(256)
void k_emit_lse_combine(const float* __restrict__ pm, const float* __restrict__ ps,
                        float* __restrict__ lse_s) {
  int s = blockIdx.x * 256 + threadIdx.x;
  float M = -3.0e38f;
  for (int c = 0; c < NCHUNK; ++c) M = fmaxf(M, pm[(size_t)c * NS + s]);
  float S = 0.f;
  for (int c = 0; c < NCHUNK; ++c)
    S += ps[(size_t)c * NS + s] * expf(pm[(size_t)c * NS + s] - M);
  lse_s[s] = M + logf(S);
}

// ---------------------------------------------------------------------------
// em_all[tb, s] = G[tb].h_emit[s] - 16*lse_s[s].  grid (128,64), 1 wave.
// ---------------------------------------------------------------------------
__global__ __launch_bounds__(32)
void k_emall(const __bf16* __restrict__ G_bf, const __bf16* __restrict__ pk_emit,
             const float* __restrict__ lse_s, float* __restrict__ em_all) {
  const int m0 = blockIdx.x * 16, nt = blockIdx.y;
  const int lane = threadIdx.x;
  v8f acc0 = {}, acc1 = {};
#pragma unroll
  for (int kt = 0; kt < 8; kt += 2) {
    acc0 = wmma_bf16(load_a_bf16row(G_bf + (size_t)m0 * EMBD + kt * 32, EMBD),
                     load_frag_packed(pk_emit + ((size_t)nt * 8 + kt) * FRAG_ELEMS), acc0);
    acc1 = wmma_bf16(load_a_bf16row(G_bf + (size_t)m0 * EMBD + (kt + 1) * 32, EMBD),
                     load_frag_packed(pk_emit + ((size_t)nt * 8 + kt + 1) * FRAG_ELEMS), acc1);
  }
  int n = lane & 15, mh = (lane >> 4) << 3;
#pragma unroll
  for (int r = 0; r < 8; ++r)
    em_all[(size_t)(m0 + mh + r) * NS + nt * 16 + n] =
        acc0[r] + acc1[r] - (float)NL * lse_s[nt * 16 + n];
}

// ---------------------------------------------------------------------------
// score0 = em_all[t=0] + priors
// ---------------------------------------------------------------------------
__global__ __launch_bounds__(256)
void k_score0(const float* __restrict__ em_all, const float* __restrict__ priors,
              float* __restrict__ out0) {
  int i = blockIdx.x * 256 + threadIdx.x;
  out0[i] = em_all[i] + priors[i & 1023];
}

// ---------------------------------------------------------------------------
// Scan prep: m_b[b] = max_s score, Pp = exp(score-m_b) packed in A-frag layout.
// ---------------------------------------------------------------------------
__global__ __launch_bounds__(256)
void k_scan_prep(const float* __restrict__ score_prev, __bf16* __restrict__ Pp,
                 float* __restrict__ m_b) {
  __shared__ float red[256];
  int b = blockIdx.x, t = threadIdx.x;
  const float* row = score_prev + (size_t)b * NS;
  float mx = -3.0e38f;
  for (int s = t; s < NS; s += 256) mx = fmaxf(mx, row[s]);
  red[t] = mx;
  __syncthreads();
  for (int off = 128; off; off >>= 1) {
    if (t < off) red[t] = fmaxf(red[t], red[t + off]);
    __syncthreads();
  }
  float M = red[0];
  int bt = b >> 4, m = b & 15;
  for (int s = t; s < NS; s += 256) {
    float p = expf(row[s] - M);
    int kt = s >> 5, kin = s & 31;
    int hi = (kin >> 3) & 1;
    int j = ((kin >> 4) << 3) | (kin & 7);
    int lane = m + (hi << 4);
    Pp[(((size_t)bt * 32 + kt) * 32 + lane) * 16 + j] = (__bf16)p;
  }
  if (t == 0) m_b[b] = M;
}

// ---------------------------------------------------------------------------
// Scan GEMM: out = em_t + m_b + c_dst + log(P @ Q).  grid (4,64), 1 wave.
// 32 K-tiles, dual accumulators, all-packed 128-bit fragment loads.
// ---------------------------------------------------------------------------
__global__ __launch_bounds__(32)
void k_scan_gemm(const __bf16* __restrict__ Pp, const __bf16* __restrict__ Qp,
                 const float* __restrict__ m_b, const float* __restrict__ cmax,
                 const float* __restrict__ em_t, float* __restrict__ out_t) {
  const int bt = blockIdx.x, nt = blockIdx.y;
  const int b0 = bt * 16, n0 = nt * 16;
  const int lane = threadIdx.x;
  const __bf16* Pbase = Pp + (size_t)bt * 32 * FRAG_ELEMS;
  v8f acc0 = {}, acc1 = {};
#pragma unroll 8
  for (int kt = 0; kt < NS / 32; kt += 2) {
    acc0 = wmma_bf16(load_frag_packed(Pbase + (size_t)kt * FRAG_ELEMS),
                     load_frag_packed(Qp + ((size_t)kt * 64 + nt) * FRAG_ELEMS), acc0);
    acc1 = wmma_bf16(load_frag_packed(Pbase + (size_t)(kt + 1) * FRAG_ELEMS),
                     load_frag_packed(Qp + ((size_t)(kt + 1) * 64 + nt) * FRAG_ELEMS), acc1);
  }
  int n = lane & 15, mh = (lane >> 4) << 3;
#pragma unroll
  for (int r = 0; r < 8; ++r) {
    int m = mh + r;
    float s = acc0[r] + acc1[r];
    out_t[(size_t)(b0 + m) * NS + n0 + n] =
        em_t[(size_t)(b0 + m) * NS + n0 + n] + m_b[b0 + m] + cmax[n0 + n] +
        logf(fmaxf(s, 1e-37f));
  }
}

// ---------------------------------------------------------------------------
// Host launcher. Input order: recursive insertion order of setup_inputs():
//  0 stories  1 story_length  2 state_emb  3 tok_emb
//  4..9 start_res{w1,b1,w2,b2,g,beta}  10 start_w  11 start_b
//  12..17 in_res  18..23 out_res  24..29 emit_res
// Workspace: ~31 MB.
// ---------------------------------------------------------------------------
extern "C" void kernel_launch(void* const* d_in, const int* in_sizes, int n_in,
                              void* d_out, int out_size, void* d_ws, size_t ws_size,
                              hipStream_t stream) {
  (void)in_sizes; (void)n_in; (void)out_size; (void)ws_size;
  const int*   stories   = (const int*)d_in[0];
  const float* state_emb = (const float*)d_in[2];
  const float* tok_emb   = (const float*)d_in[3];
  const float* start_w   = (const float*)d_in[10];
  const float* start_b   = (const float*)d_in[11];

  char* ws = (char*)d_ws;
  size_t off = 0;
  auto carve = [&](size_t bytes) -> void* {
    void* p = ws + off;
    off += (bytes + 255) & ~(size_t)255;
    return p;
  };
  float*  h_start  = (float*)carve((size_t)NS * EMBD * 4);
  float*  h_in     = (float*)carve((size_t)NS * EMBD * 4);
  float*  h_out    = (float*)carve((size_t)NS * EMBD * 4);
  float*  h_emit   = (float*)carve((size_t)NS * EMBD * 4);
  __bf16* hb_start = (__bf16*)carve((size_t)NS * EMBD * 2);
  __bf16* hb_in    = (__bf16*)carve((size_t)NS * EMBD * 2);
  __bf16* hb_out   = (__bf16*)carve((size_t)NS * EMBD * 2);
  __bf16* hb_emit  = (__bf16*)carve((size_t)NS * EMBD * 2);
  __bf16* hb_state = (__bf16*)carve((size_t)NS * EMBD * 2);   // state_emb bf16 rows
  __bf16* pk_w[8];                                            // packed weight tiles
  for (int i = 0; i < 8; ++i) pk_w[i] = (__bf16*)carve((size_t)EMBD * EMBD * 2);
  __bf16* pk_out   = (__bf16*)carve((size_t)NS * EMBD * 2);   // h_out^T packed
  __bf16* pk_emit  = (__bf16*)carve((size_t)NS * EMBD * 2);   // h_emit^T packed
  __bf16* pk_tok   = (__bf16*)carve((size_t)NV * EMBD * 2);   // tok_emb^T packed
  float*  priors   = (float*)carve(NS * 4);
  float*  logits   = (float*)carve((size_t)NS * NS * 4);
  float*  row_lse  = (float*)carve(NS * 4);
  float*  cmax     = (float*)carve(NS * 4);
  __bf16* Qp       = (__bf16*)carve((size_t)NS * NS * 2);
  __bf16* G_bf     = (__bf16*)carve((size_t)NT * NB * EMBD * 2);
  float*  pm       = (float*)carve((size_t)NCHUNK * NS * 4);
  float*  ps       = (float*)carve((size_t)NCHUNK * NS * 4);
  float*  lse_s    = (float*)carve(NS * 4);
  float*  em_all   = (float*)carve((size_t)NT * NB * NS * 4);
  __bf16* Pp       = (__bf16*)carve((size_t)NB * NS * 2);
  float*  m_b      = (float*)carve(NB * 4);

  // One-time operand preparation.
  k_cvt_bf16<<<(NS * EMBD) / 256, 256, 0, stream>>>(state_emb, hb_state);
  const int wbase[4] = {4, 12, 18, 24};
  for (int i = 0; i < 4; ++i) {
    k_pack_bT<<<EMBD, 256, 0, stream>>>((const float*)d_in[wbase[i]],     pk_w[2 * i]);     // w1
    k_pack_bT<<<EMBD, 256, 0, stream>>>((const float*)d_in[wbase[i] + 2], pk_w[2 * i + 1]); // w2
  }

  float*  hf[4] = {h_start, h_in, h_out, h_emit};
  __bf16* hb[4] = {hb_start, hb_in, hb_out, hb_emit};
  for (int i = 0; i < 4; ++i) {
    int base = wbase[i];
    k_residual<<<NS / 16, 32, 0, stream>>>(
        hb_state, pk_w[2 * i], (const float*)d_in[base + 1],
        pk_w[2 * i + 1],       (const float*)d_in[base + 3],
        (const float*)d_in[base + 4], (const float*)d_in[base + 5],
        hf[i], hb[i]);
  }

  k_pack_bT<<<NS, 256, 0, stream>>>(h_out,  pk_out);
  k_pack_bT<<<NS, 256, 0, stream>>>(h_emit, pk_emit);
  k_pack_bT<<<NV, 256, 0, stream>>>(tok_emb, pk_tok);

  k_priors<<<1, 1024, 0, stream>>>(h_start, start_w, start_b, priors);

  k_trans_logits<<<dim3(NS / 16, NS / 16), 32, 0, stream>>>(hb_in, pk_out, logits);
  k_row_lse<<<NS, 256, 0, stream>>>(logits, row_lse);
  k_col_max<<<NS / 256, 256, 0, stream>>>(logits, row_lse, cmax);
  k_qexp<<<(NS * NS) / 256, 256, 0, stream>>>(logits, row_lse, cmax, Qp);

  k_gather<<<NT * NB, 256, 0, stream>>>(stories, tok_emb, G_bf);
  k_emit_lse_partial<<<dim3(NS / 16, NCHUNK), 32, 0, stream>>>(hb_emit, pk_tok, pm, ps);
  k_emit_lse_combine<<<NS / 256, 256, 0, stream>>>(pm, ps, lse_s);
  k_emall<<<dim3((NT * NB) / 16, NS / 16), 32, 0, stream>>>(G_bf, pk_emit, lse_s, em_all);

  float* out = (float*)d_out;   // (T, B, S)
  k_score0<<<(NB * NS) / 256, 256, 0, stream>>>(em_all, priors, out);

  for (int t = 1; t < NT; ++t) {
    const float* prev = out + (size_t)(t - 1) * NB * NS;
    k_scan_prep<<<NB, 256, 0, stream>>>(prev, Pp, m_b);
    k_scan_gemm<<<dim3(NB / 16, NS / 16), 32, 0, stream>>>(
        Pp, Qp, m_b, cmax, em_all + (size_t)t * NB * NS, out + (size_t)t * NB * NS);
  }
}
